// GalaxyLSSBackbone_6992206758101
// MI455X (gfx1250) — matrix operand order repaired and checked by reading, not compiled, not run on hardware
//
#include <hip/hip_runtime.h>
#include <hip/hip_bf16.h>
#include <math.h>

// ---------------- problem constants ----------------
#define SD 32
#define VD 32
#define NLAYERS 3
#define IN_DIM 99          // 2*SD + VD + 3
#define IN_PAD 128         // padded K for WMMA (4 chunks of 32)
#define OUT_DIM 64         // SD + VD
#define A_STRIDE 136       // LDS row stride (halves) for msg tile, bank-conflict pad
#define W1_STRIDE 136      // LDS row stride (halves) for W1
#define H_STRIDE 72        // LDS row stride (halves) for hidden tile / W2
#define WAVES_PER_WG 4
#define WG_SIZE (WAVES_PER_WG * 32)

typedef __attribute__((ext_vector_type(16))) _Float16 v16h;
typedef __attribute__((ext_vector_type(8)))  float    v8f;

union FragU {
    v16h v;
    _Float16 h[16];
    unsigned u[8];
};

// Wave-local LDS ordering: DS ops are in-order within a wave; we only need to
// stop the compiler from reordering across the phase boundary, plus drain DS.
__device__ inline void wave_lds_sync() {
    __builtin_amdgcn_wave_barrier();
    asm volatile("s_wait_dscnt 0" ::: "memory");
    __builtin_amdgcn_wave_barrier();
}

// A-fragment (16x32 f16, M = edge row) from an LDS tile with row stride `rs` halves.
// Layout (ISA 7.12.2): lanes 0-15: M=lane, halves = K{0..7,16..23}; lanes 16-31: +8.
__device__ inline v16h load_a_frag(const _Float16* base, int rs, int kbase, int lane) {
    FragU f;
    const int m    = lane & 15;
    const int koff = (lane & 16) ? 8 : 0;
    const _Float16* row = base + m * rs + kbase + koff;
#pragma unroll
    for (int p = 0; p < 8; ++p) {
        const int kp = (p < 4) ? (2 * p) : (16 + 2 * (p - 4));
        f.u[p] = *(const unsigned*)(row + kp);
    }
    return f.v;
}

// B-fragment (32x16 f16, B[k][n] = W[n][k]) from LDS weight tile stored row-major
// as W[n][k] with row stride `rs` halves. Lanes 0-15: K{0..7,16..23}; lanes 16-31: +8.
__device__ inline v16h load_b_frag(const _Float16* base, int rs, int nbase, int kbase, int lane) {
    FragU f;
    const int n  = nbase + (lane & 15);
    const int ks = kbase + ((lane & 16) ? 8 : 0);
    const unsigned* p = (const unsigned*)(base + n * rs + ks);
#pragma unroll
    for (int i = 0; i < 8; ++i) f.u[i] = p[(i < 4) ? i : (i + 4)];
    return f.v;
}

__device__ inline float silu_f(float x) {
    return x / (1.0f + __expf(-x));
}

// ---------------- encoder ----------------
__global__ void encode_kernel(const float* __restrict__ redshift,
                              const float* __restrict__ shapes,
                              const float* __restrict__ esw,
                              const float* __restrict__ esb,
                              const float* __restrict__ evw,
                              float* __restrict__ h, int total) {
    int i = blockIdx.x * blockDim.x + threadIdx.x;
    if (i >= total) return;
    int node = i >> 6;
    int c = i & 63;
    float v;
    if (c < SD) {
        v = redshift[node] * esw[c] + esb[c];
        v = v > 0.0f ? v : 0.0f;                       // relu
    } else {
        int j = c - SD;
        v = shapes[node * 2] * evw[j * 2] + shapes[node * 2 + 1] * evw[j * 2 + 1];
    }
    h[i] = v;
}

__global__ void copy_kernel(const float* __restrict__ in, float* __restrict__ out, int n) {
    int i = blockIdx.x * blockDim.x + threadIdx.x;
    if (i < n) out[i] = in[i];
}

__global__ void output_kernel(const float* __restrict__ h, float* __restrict__ out, int N) {
    int i = blockIdx.x * blockDim.x + threadIdx.x;
    if (i >= N * 64) return;
    int node = i >> 6;
    int c = i & 63;
    if (c < SD) out[node * SD + c] = h[i];                         // h_s
    else        out[(size_t)N * SD + node * VD + (c - SD)] = h[i]; // h_v
}

// ---------------- edge message-passing layer (WMMA) ----------------
__global__ __launch_bounds__(WG_SIZE) void edge_kernel(
        const float* __restrict__ h_in,        // [N][64] (cols 0..31 = h_s, 32..63 = h_v)
        float* __restrict__ h_out,             // [N][64], pre-initialized with h_in (residual)
        const long long* __restrict__ eidx,    // [2][E] int64
        const float* __restrict__ pos,         // [N][2]
        const float* __restrict__ W1,          // [64][99] (layer slice)
        const float* __restrict__ b1,          // [64]
        const float* __restrict__ W2,          // [64][64]
        const float* __restrict__ b2,          // [64]
        int nTiles, long long E) {

    __shared__ __align__(16) _Float16 sW1[OUT_DIM * W1_STRIDE];
    __shared__ __align__(16) _Float16 sW2[OUT_DIM * H_STRIDE];
    __shared__ __align__(16) float    sB1[OUT_DIM];
    __shared__ __align__(16) float    sB2[OUT_DIM];
    __shared__ __align__(16) _Float16 sA[WAVES_PER_WG][16 * A_STRIDE];
    __shared__ __align__(16) _Float16 sH[WAVES_PER_WG][16 * H_STRIDE];
    __shared__ __align__(16) int      sDst[WAVES_PER_WG][16];

    const int tid  = threadIdx.x;
    const int wave = tid >> 5;
    const int lane = tid & 31;

    // ---- stage weights (f16) + biases, once per workgroup ----
    for (int idx = tid; idx < OUT_DIM * IN_PAD; idx += WG_SIZE) {
        int n = idx >> 7;          // /128
        int k = idx & 127;
        sW1[n * W1_STRIDE + k] = (k < IN_DIM) ? (_Float16)W1[n * IN_DIM + k] : (_Float16)0.0f;
    }
    for (int idx = tid; idx < OUT_DIM * OUT_DIM; idx += WG_SIZE) {
        int n = idx >> 6;
        int k = idx & 63;
        sW2[n * H_STRIDE + k] = (_Float16)W2[n * OUT_DIM + k];
    }
    if (tid < OUT_DIM)       sB1[tid] = b1[tid];
    else if (tid < 2 * OUT_DIM) sB2[tid - OUT_DIM] = b2[tid - OUT_DIM];
    __syncthreads();

    _Float16* Aw = sA[wave];
    _Float16* Hw = sH[wave];
    int*      Dw = sDst[wave];

    const int m0  = (lane & 16) ? 8 : 0;     // C/D fragment row base for this lane half
    const int nn  = lane & 15;               // C/D fragment column-within-tile

    const int waveStride = gridDim.x * WAVES_PER_WG;
    for (int tile = blockIdx.x * WAVES_PER_WG + wave; tile < nTiles; tile += waveStride) {
        const long long ebase = (long long)tile * 16;

        // ---- gather: geo features + dst indices (lanes 0..15, one edge each) ----
        if (lane < 16) {
            long long e = ebase + lane;
            if (e >= E) e = 0;
            const int s = (int)eidx[e];
            const int d = (int)eidx[E + e];
            Dw[lane] = d;
            const float dx = pos[2 * s] - pos[2 * d];
            const float dy = pos[2 * s + 1] - pos[2 * d + 1];
            const float r2 = dx * dx + dy * dy;
            const float dist = sqrtf(r2) + 1e-6f;
            float c2 = 1.0f, s2 = 0.0f;                 // alpha==beta==0 -> 2*phi identities
            if (r2 > 0.0f) {
                const float inv = 1.0f / r2;
                c2 = (dx * dx - dy * dy) * inv;
                s2 = 2.0f * dx * dy * inv;
            }
            _Float16* row = Aw + lane * A_STRIDE;
            row[96] = (_Float16)dist;
            row[97] = (_Float16)c2;
            row[98] = (_Float16)s2;
#pragma unroll
            for (int k = IN_DIM; k < IN_PAD; ++k) row[k] = (_Float16)0.0f;
        }

        // ---- gather: node features, coalesced 32-lane rows ----
        for (int r = 0; r < 16; ++r) {
            long long e = ebase + r;
            if (e >= E) e = 0;
            const int s = (int)eidx[e];      // uniform across the wave -> scalar load
            const int d = (int)eidx[E + e];
            const float* hs = h_in + (size_t)s * 64;
            const float* hd = h_in + (size_t)d * 64;
            _Float16* row = Aw + r * A_STRIDE;
            row[lane]      = (_Float16)hs[lane];        // h_s[src]
            row[32 + lane] = (_Float16)hd[lane];        // h_s[dst]
            row[64 + lane] = (_Float16)hs[32 + lane];   // v_rot == h_v[src] (rot == 0)
        }
        wave_lds_sync();

        // ---- layer 1: hid(16x64) = silu(msg(16x128) @ W1^T + b1) ----
#pragma unroll
        for (int nt = 0; nt < 4; ++nt) {
            const float bv = sB1[nt * 16 + nn];
            v8f acc = {bv, bv, bv, bv, bv, bv, bv, bv};
#pragma unroll
            for (int kc = 0; kc < 4; ++kc) {
                v16h a = load_a_frag(Aw, A_STRIDE, kc * 32, lane);
                v16h b = load_b_frag(sW1, W1_STRIDE, nt * 16, kc * 32, lane);
                acc = __builtin_amdgcn_wmma_f32_16x16x32_f16(
                        false, a, false, b, (short)0, acc, false, false);
            }
            const int col = nt * 16 + nn;
#pragma unroll
            for (int r = 0; r < 8; ++r) {
                Hw[(r + m0) * H_STRIDE + col] = (_Float16)silu_f(acc[r]);
            }
        }
        wave_lds_sync();

        // ---- layer 2: raw(16x64) = hid @ W2^T + b2 ; scatter-add to dst nodes ----
#pragma unroll
        for (int nt = 0; nt < 4; ++nt) {
            const float bv = sB2[nt * 16 + nn];
            v8f acc = {bv, bv, bv, bv, bv, bv, bv, bv};
#pragma unroll
            for (int kc = 0; kc < 2; ++kc) {
                v16h a = load_a_frag(Hw, H_STRIDE, kc * 32, lane);
                v16h b = load_b_frag(sW2, H_STRIDE, nt * 16, kc * 32, lane);
                acc = __builtin_amdgcn_wmma_f32_16x16x32_f16(
                        false, a, false, b, (short)0, acc, false, false);
            }
            const int col = nt * 16 + nn;   // 0..31 -> h_s part, 32..63 -> h_v part
#pragma unroll
            for (int r = 0; r < 8; ++r) {
                const int M = r + m0;
                if (ebase + M < E) {
                    const int dn = Dw[M];
                    atomicAdd(&h_out[(size_t)dn * 64 + col], acc[r]);
                }
            }
        }
        __builtin_amdgcn_wave_barrier();   // keep next tile's DS writes behind this tile's reads
    }
}

// ---------------- launch ----------------
extern "C" void kernel_launch(void* const* d_in, const int* in_sizes, int n_in,
                              void* d_out, int out_size, void* d_ws, size_t ws_size,
                              hipStream_t stream) {
    const float*     pos      = (const float*)d_in[0];
    const float*     redshift = (const float*)d_in[1];
    const float*     shapes   = (const float*)d_in[2];
    const long long* eidx     = (const long long*)d_in[3];   // int64 per reference
    const float*     esw      = (const float*)d_in[4];
    const float*     esb      = (const float*)d_in[5];
    const float*     evw      = (const float*)d_in[6];
    const float*     W1       = (const float*)d_in[7];       // [L][64][99]
    const float*     b1       = (const float*)d_in[8];       // [L][64]
    const float*     W2       = (const float*)d_in[9];       // [L][64][64]
    const float*     b2       = (const float*)d_in[10];      // [L][64]
    float*           out      = (float*)d_out;

    const int       N = in_sizes[1];                 // redshift: N elements
    const long long E = (long long)in_sizes[3] / 2;  // edge_index: 2*E
    const int total   = N * 64;

    float* bufA = (float*)d_ws;
    float* bufB = bufA + (size_t)total;

    encode_kernel<<<(total + 255) / 256, 256, 0, stream>>>(redshift, shapes, esw, esb, evw,
                                                           bufA, total);

    const int nTiles = (int)((E + 15) / 16);
    int nWG = (nTiles + WAVES_PER_WG - 1) / WAVES_PER_WG;
    if (nWG > 2048) nWG = 2048;

    for (int l = 0; l < NLAYERS; ++l) {
        copy_kernel<<<(total + 255) / 256, 256, 0, stream>>>(bufA, bufB, total);
        edge_kernel<<<nWG, WG_SIZE, 0, stream>>>(
            bufA, bufB, eidx, pos,
            W1 + (size_t)l * OUT_DIM * IN_DIM,
            b1 + (size_t)l * OUT_DIM,
            W2 + (size_t)l * OUT_DIM * OUT_DIM,
            b2 + (size_t)l * OUT_DIM,
            nTiles, E);
        float* t = bufA; bufA = bufB; bufB = t;
    }

    output_kernel<<<(total + 255) / 256, 256, 0, stream>>>(bufA, out, N);
}
